// BakaAttention_51960514347180
// MI455X (gfx1250) — compile-verified, matrix operand order checked by reading, not run on hardware
//
#include <hip/hip_runtime.h>
#include <hip/hip_bf16.h>
#include <math.h>

// ---------------------------------------------------------------------------
// Types
// ---------------------------------------------------------------------------
typedef __attribute__((ext_vector_type(16))) _Float16 v16h;
typedef __attribute__((ext_vector_type(8)))  _Float16 vh8;
typedef __attribute__((ext_vector_type(4)))  _Float16 vh4;
typedef __attribute__((ext_vector_type(8)))  float    v8f;
typedef __attribute__((ext_vector_type(4)))  float    vf4;
typedef __attribute__((ext_vector_type(4)))  int      vi4;

#define B_  4
#define T_  1024
#define P_  1024
#define S_  2048
#define D_  2048
#define H_  16
#define HD_ 128
#define SCALE_ 0.08838834764831845f   // 1/sqrt(128)

// ---------------------------------------------------------------------------
// Async copy (CDNA5 global_load_async_to_lds) with sync fallback
// ---------------------------------------------------------------------------
#if defined(__has_builtin)
#if __has_builtin(__builtin_amdgcn_global_load_async_to_lds_b128)
#define HAVE_ASYNC 1
#endif
#endif

#if defined(HAVE_ASYNC)
typedef __attribute__((address_space(1))) vi4 gvi4_t;   // global int4
typedef __attribute__((address_space(3))) vi4 lvi4_t;   // LDS int4
#endif

__device__ __forceinline__ void async_copy16(const void* g, void* l) {
#if defined(HAVE_ASYNC)
  __builtin_amdgcn_global_load_async_to_lds_b128(
      (gvi4_t*)g, (lvi4_t*)l, 0, 0);
#else
  *(vf4*)l = *(const vf4*)g;
#endif
}

__device__ __forceinline__ void async_wait() {
#if defined(HAVE_ASYNC)
#if __has_builtin(__builtin_amdgcn_s_wait_asynccnt)
  __builtin_amdgcn_s_wait_asynccnt(0);
#else
  asm volatile("s_wait_asynccnt 0" ::: "memory");
#endif
#endif
}

// ---------------------------------------------------------------------------
// WMMA helpers
// ---------------------------------------------------------------------------
__device__ __forceinline__ v8f wmma_f16(v16h a, v16h b, v8f c) {
  return __builtin_amdgcn_wmma_f32_16x16x32_f16(false, a, false, b,
                                                (short)0, c, false, false);
}

// Fragment = two contiguous 16-byte chunks (ISA 16-bit A/B layout)
__device__ __forceinline__ v16h ld_frag(const _Float16* p0, const _Float16* p1) {
  union { v16h f; vh8 h[2]; } u;
  u.h[0] = *(const vh8*)p0;
  u.h[1] = *(const vh8*)p1;
  return u.f;
}

// ---------------------------------------------------------------------------
// K0: RoPE cos/sin table  [S_][HD_]
// ---------------------------------------------------------------------------
__global__ void k_rope_table(float* __restrict__ cosT, float* __restrict__ sinT) {
  int idx = blockIdx.x * 256 + threadIdx.x;   // S_*HD_ = 262144
  int pos = idx >> 7;
  int hd  = idx & 127;
  float e   = (float)(hd & ~1) * (1.0f / 128.0f);
  float inv = powf(10000.0f, -e);
  float a   = (float)pos * inv;
  cosT[idx] = cosf(a);
  sinT[idx] = sinf(a);
}

// ---------------------------------------------------------------------------
// K1: past_k -> RoPE -> Kc[b,h,s,hd] (f16); past_v -> Vt[b,h,hd,s] (f16, transposed)
// grid (P_/128, H_, B_), 256 threads
// ---------------------------------------------------------------------------
__global__ __launch_bounds__(256)
void k_past(const float* __restrict__ past_k, const float* __restrict__ past_v,
            const float* __restrict__ cosT, const float* __restrict__ sinT,
            _Float16* __restrict__ Kc, _Float16* __restrict__ Vt) {
  __shared__ __align__(16) _Float16 ct[128 * 136];
  int b = blockIdx.z, h = blockIdx.y, s0 = blockIdx.x * 128;
  int tid = threadIdx.x;

#pragma unroll
  for (int r = 0; r < 16; ++r) {
    int idx = tid + r * 256;            // 0..4095
    int sl  = idx >> 5;                 // local s row
    int seg = idx & 31;                 // float4 segment
    int hd  = seg * 4;
    int s   = s0 + sl;
    size_t src = (((size_t)(b * T_ + s) * H_ + h) << 7) + hd;

    // K with RoPE (pairs are inside the float4: no cross-lane traffic)
    vf4 kv = *(const vf4*)(past_k + src);
    float c0 = cosT[s * HD_ + hd],     sn0 = sinT[s * HD_ + hd];
    float c1 = cosT[s * HD_ + hd + 2], sn1 = sinT[s * HD_ + hd + 2];
    vh4 ko;
    ko[0] = (_Float16)(kv.x * c0 - kv.y * sn0);
    ko[1] = (_Float16)(kv.y * c0 + kv.x * sn0);
    ko[2] = (_Float16)(kv.z * c1 - kv.w * sn1);
    ko[3] = (_Float16)(kv.w * c1 + kv.z * sn1);
    *(vh4*)(Kc + (((size_t)(b * H_ + h) * S_ + s) << 7) + hd) = ko;

    // V -> LDS staging (row-major [s][hd])
    vf4 vv = *(const vf4*)(past_v + src);
    _Float16* cp = ct + sl * 136 + hd;
    cp[0] = (_Float16)vv.x; cp[1] = (_Float16)vv.y;
    cp[2] = (_Float16)vv.z; cp[3] = (_Float16)vv.w;
  }
  __syncthreads();

  // transposed, coalesced write-out: Vt[b,h,hd, s0..s0+127]
#pragma unroll
  for (int q = 0; q < 8; ++q) {
    int idx = tid + q * 256;            // 0..2047
    int hd  = idx >> 4;
    int ck  = idx & 15;                 // 8-half chunk along s
    vh8 tv;
#pragma unroll
    for (int c = 0; c < 8; ++c) tv[c] = ct[(ck * 8 + c) * 136 + hd];
    *(vh8*)(Vt + ((size_t)(b * H_ + h) * HD_ + hd) * S_ + s0 + ck * 8) = tv;
  }
}

// ---------------------------------------------------------------------------
// K2: fused QKV projection GEMM  C = x @ W{q,k,v}  (f16 WMMA, f32 accum)
// grid (D_/128, (B_*T_)/128, 3), 256 threads. Epilogue: RoPE + layout scatter.
// ---------------------------------------------------------------------------
__global__ __launch_bounds__(256)
void k_qkv(const float* __restrict__ x,
           const float* __restrict__ Wq, const float* __restrict__ Wk,
           const float* __restrict__ Wv,
           const float* __restrict__ cosT, const float* __restrict__ sinT,
           _Float16* __restrict__ Qc, _Float16* __restrict__ Kc,
           _Float16* __restrict__ Vt) {
  __shared__ __align__(16) _Float16 smem[128 * 136];  // As(128x40) Bs(128x40) / ct(128x136)
  _Float16* As = smem;
  _Float16* Bs = smem + 128 * 40;

  int which = blockIdx.z;
  const float* W = (which == 0) ? Wq : (which == 1) ? Wk : Wv;
  int m0 = blockIdx.y * 128, n0 = blockIdx.x * 128;
  int tid = threadIdx.x;
  int lane = tid & 31, wid = tid >> 5;
  int wm = wid & 3, wn = wid >> 2;      // 4x2 wave grid -> 32x64 per wave
  int lh = lane >> 4, l16 = lane & 15;

  v8f acc[2][4] = {};

  for (int kt = 0; kt < D_; kt += 32) {
    // A tile: x[m0..+127][kt..+31] fp32 -> f16 LDS (row stride 40)
#pragma unroll
    for (int r = 0; r < 4; ++r) {
      int idx = tid + r * 256;          // 0..1023
      int row = idx >> 3, seg = idx & 7;
      vf4 v = *(const vf4*)(x + (size_t)(m0 + row) * D_ + kt + seg * 4);
      vh4 hv; hv[0]=(_Float16)v.x; hv[1]=(_Float16)v.y; hv[2]=(_Float16)v.z; hv[3]=(_Float16)v.w;
      *(vh4*)(As + row * 40 + seg * 4) = hv;
    }
    // B tile: W[kt..+31][n0..+127] fp32 -> transposed f16 LDS Bs[n][k]
#pragma unroll
    for (int r = 0; r < 4; ++r) {
      int idx = tid + r * 256;
      int kr = idx >> 5, seg = idx & 31;
      vf4 v = *(const vf4*)(W + (size_t)(kt + kr) * D_ + n0 + seg * 4);
      int nb = seg * 4;
      Bs[(nb + 0) * 40 + kr] = (_Float16)v.x;
      Bs[(nb + 1) * 40 + kr] = (_Float16)v.y;
      Bs[(nb + 2) * 40 + kr] = (_Float16)v.z;
      Bs[(nb + 3) * 40 + kr] = (_Float16)v.w;
    }
    __syncthreads();

    v16h a[2], bf[4];
#pragma unroll
    for (int i = 0; i < 2; ++i) {
      int row = wm * 32 + i * 16 + l16;
      a[i] = ld_frag(As + row * 40 + lh * 8, As + row * 40 + 16 + lh * 8);
    }
#pragma unroll
    for (int j = 0; j < 4; ++j) {
      int row = wn * 64 + j * 16 + l16;
      bf[j] = ld_frag(Bs + row * 40 + lh * 8, Bs + row * 40 + 16 + lh * 8);
    }
#pragma unroll
    for (int i = 0; i < 2; ++i)
#pragma unroll
      for (int j = 0; j < 4; ++j)
        acc[i][j] = wmma_f16(a[i], bf[j], acc[i][j]);
    __syncthreads();
  }

  // Epilogue
  int h = n0 >> 7;                      // BN==HD_: one head per block
#pragma unroll
  for (int i = 0; i < 2; ++i) {
#pragma unroll
    for (int j = 0; j < 4; ++j) {
#pragma unroll
      for (int r = 0; r < 8; ++r) {
        int ml = wm * 32 + i * 16 + r + 8 * lh;
        int nl = wn * 64 + j * 16 + l16;
        int m = m0 + ml;
        int t = m & (T_ - 1), b = m >> 10;
        int hd = nl & 127;
        float val = acc[i][j][r];
        if (which == 2) {
          smem[ml * 136 + nl] = (_Float16)val;      // stage for V transpose
        } else {
          int pos = t + P_;                          // q: offset P; k_new: index P+t
          float cv = cosT[pos * HD_ + hd];
          float sv = sinT[pos * HD_ + hd];
          float partner = __shfl_xor(val, 1);        // hd parity == lane parity
          float rot = (hd & 1) ? partner : -partner;
          _Float16 o = (_Float16)(val * cv + rot * sv);
          if (which == 0)
            Qc[(((size_t)(b * H_ + h) * T_ + t) << 7) + hd] = o;
          else
            Kc[(((size_t)(b * H_ + h) * S_ + P_ + t) << 7) + hd] = o;
        }
      }
    }
  }
  if (which == 2) {
    __syncthreads();
    int b = m0 >> 10, t0 = m0 & (T_ - 1);
#pragma unroll
    for (int q = 0; q < 8; ++q) {
      int idx = tid + q * 256;          // 0..2047
      int hd = idx >> 4, ck = idx & 15;
      vh8 tv;
#pragma unroll
      for (int c = 0; c < 8; ++c) tv[c] = smem[(ck * 8 + c) * 136 + hd];
      *(vh8*)(Vt + ((size_t)(b * H_ + h) * HD_ + hd) * S_ + P_ + t0 + ck * 8) = tv;
    }
  }
}

// ---------------------------------------------------------------------------
// K3: causal flash attention. grid (T_/128, H_, B_), 256 threads (8 waves).
// Each wave owns 16 q rows; block streams 32-row K/V tiles through LDS.
// ---------------------------------------------------------------------------
__global__ __launch_bounds__(256)
void k_attn(const _Float16* __restrict__ Qc, const _Float16* __restrict__ Kc,
            const _Float16* __restrict__ Vt, _Float16* __restrict__ Yc) {
  __shared__ __align__(16) _Float16 Ks[32 * 136];    // [s][hd]
  __shared__ __align__(16) _Float16 Vs[128 * 40];    // [hd][s]
  __shared__ __align__(16) _Float16 Ps[8 * 16 * 40]; // per-wave P stage [m][s]

  int b = blockIdx.z, h = blockIdx.y, t0 = blockIdx.x * 128;
  int tid = threadIdx.x;
  int lane = tid & 31, wid = tid >> 5;
  int lh = lane >> 4, l16 = lane & 15;

  size_t baseQ = ((size_t)(b * H_ + h) * T_) << 7;
  size_t baseK = ((size_t)(b * H_ + h) * S_) << 7;
  size_t baseV = (size_t)(b * H_ + h) * HD_ * S_;

  // Q fragments held in registers across the whole s loop
  v16h aQ[4];
  {
    const _Float16* qp = Qc + baseQ + ((size_t)(t0 + wid * 16 + l16) << 7);
#pragma unroll
    for (int kt = 0; kt < 4; ++kt)
      aQ[kt] = ld_frag(qp + kt * 32 + lh * 8, qp + kt * 32 + 16 + lh * 8);
  }

  float mrow[8], lrow[8];
  v8f o[8] = {};
#pragma unroll
  for (int r = 0; r < 8; ++r) { mrow[r] = -INFINITY; lrow[r] = 0.0f; }

  int nIter = (t0 + 127 + P_) / 32 + 1;
  for (int it = 0; it < nIter; ++it) {
    int s0 = it * 32;
    // K tile: 32 x 128 halves (rows 256B), async to LDS (row stride 272B)
#pragma unroll
    for (int r = 0; r < 2; ++r) {
      int idx = tid + r * 256;          // 0..511
      int row = idx >> 4, ck = idx & 15;
      async_copy16(Kc + baseK + ((size_t)(s0 + row) << 7) + ck * 8,
                   Ks + row * 136 + ck * 8);
    }
    // Vt tile: 128 x 32 halves (rows 64B), async to LDS (row stride 80B)
#pragma unroll
    for (int r = 0; r < 2; ++r) {
      int idx = tid + r * 256;
      int hd = idx >> 2, ck = idx & 3;
      async_copy16(Vt + baseV + (size_t)hd * S_ + s0 + ck * 8,
                   Vs + hd * 40 + ck * 8);
    }
    async_wait();
    __syncthreads();

    // scores = Q(16xHd) . K^T(Hdx32), 4 K-steps x 2 N-subtiles
    v8f sc[2] = {};
#pragma unroll
    for (int kt = 0; kt < 4; ++kt) {
#pragma unroll
      for (int j = 0; j < 2; ++j) {
        int row = j * 16 + l16;
        v16h bk = ld_frag(Ks + row * 136 + kt * 32 + lh * 8,
                          Ks + row * 136 + kt * 32 + 16 + lh * 8);
        sc[j] = wmma_f16(aQ[kt], bk, sc[j]);
      }
    }

    // mask + online softmax (16-lane shuffle reductions per C-row)
    float alpha[8];
    int tqb = t0 + wid * 16;
#pragma unroll
    for (int r = 0; r < 8; ++r) {
      int m = r + 8 * lh;
      int tq = tqb + m;
      float v0 = sc[0][r] * SCALE_, v1 = sc[1][r] * SCALE_;
      if (s0 + l16 > tq + P_)      v0 = -3.0e38f;
      if (s0 + 16 + l16 > tq + P_) v1 = -3.0e38f;
      float cand = fmaxf(v0, v1);
      cand = fmaxf(cand, __shfl_xor(cand, 1));
      cand = fmaxf(cand, __shfl_xor(cand, 2));
      cand = fmaxf(cand, __shfl_xor(cand, 4));
      cand = fmaxf(cand, __shfl_xor(cand, 8));
      float mnew = fmaxf(mrow[r], cand);
      float al = __expf(mrow[r] - mnew);
      float p0 = __expf(v0 - mnew), p1 = __expf(v1 - mnew);
      float rs = p0 + p1;
      rs += __shfl_xor(rs, 1);
      rs += __shfl_xor(rs, 2);
      rs += __shfl_xor(rs, 4);
      rs += __shfl_xor(rs, 8);
      lrow[r] = lrow[r] * al + rs;
      mrow[r] = mnew;
      alpha[r] = al;
      Ps[(wid * 16 + m) * 40 + l16]      = (_Float16)p0;
      Ps[(wid * 16 + m) * 40 + 16 + l16] = (_Float16)p1;
    }
#pragma unroll
    for (int j = 0; j < 8; ++j)
#pragma unroll
      for (int r = 0; r < 8; ++r) o[j][r] *= alpha[r];

    asm volatile("s_wait_dscnt 0" ::: "memory");   // same-wave LDS RAW (C->A relayout)

    // O += P(16x32) . V(32xHd)
    const _Float16* pw = Ps + (wid * 16 + l16) * 40;
    v16h aP = ld_frag(pw + lh * 8, pw + 16 + lh * 8);
#pragma unroll
    for (int j = 0; j < 8; ++j) {
      int row = j * 16 + l16;
      v16h bv = ld_frag(Vs + row * 40 + lh * 8, Vs + row * 40 + 16 + lh * 8);
      o[j] = wmma_f16(aP, bv, o[j]);
    }
    __syncthreads();
  }

  // normalize and write Y[b,t,h*128+hd] (f16)
#pragma unroll
  for (int r = 0; r < 8; ++r) {
    float inv = 1.0f / lrow[r];
    int t = t0 + wid * 16 + r + 8 * lh;
#pragma unroll
    for (int j = 0; j < 8; ++j) {
      Yc[(size_t)(b * T_ + t) * D_ + h * HD_ + j * 16 + l16] =
          (_Float16)(o[j][r] * inv);
    }
  }
}

// ---------------------------------------------------------------------------
// K4: output projection  out = Y(f16) @ Wo(fp32->f16), fp32 result
// grid (D_/128, (B_*T_)/128), 256 threads
// ---------------------------------------------------------------------------
__global__ __launch_bounds__(256)
void k_oproj(const _Float16* __restrict__ Yc, const float* __restrict__ Wo,
             float* __restrict__ out) {
  __shared__ __align__(16) _Float16 As[128 * 40];
  __shared__ __align__(16) _Float16 Bs[128 * 40];
  int m0 = blockIdx.y * 128, n0 = blockIdx.x * 128;
  int tid = threadIdx.x;
  int lane = tid & 31, wid = tid >> 5;
  int wm = wid & 3, wn = wid >> 2;
  int lh = lane >> 4, l16 = lane & 15;

  v8f acc[2][4] = {};

  for (int kt = 0; kt < D_; kt += 32) {
    // A tile: already f16 -> async copy straight into LDS
#pragma unroll
    for (int r = 0; r < 2; ++r) {
      int idx = tid + r * 256;          // 0..511
      int row = idx >> 2, ck = idx & 3;
      async_copy16(Yc + (size_t)(m0 + row) * D_ + kt + ck * 8,
                   As + row * 40 + ck * 8);
    }
    // B tile: Wo fp32 -> transposed f16
#pragma unroll
    for (int r = 0; r < 4; ++r) {
      int idx = tid + r * 256;
      int kr = idx >> 5, seg = idx & 31;
      vf4 v = *(const vf4*)(Wo + (size_t)(kt + kr) * D_ + n0 + seg * 4);
      int nb = seg * 4;
      Bs[(nb + 0) * 40 + kr] = (_Float16)v.x;
      Bs[(nb + 1) * 40 + kr] = (_Float16)v.y;
      Bs[(nb + 2) * 40 + kr] = (_Float16)v.z;
      Bs[(nb + 3) * 40 + kr] = (_Float16)v.w;
    }
    async_wait();
    __syncthreads();

    v16h a[2], bf[4];
#pragma unroll
    for (int i = 0; i < 2; ++i) {
      int row = wm * 32 + i * 16 + l16;
      a[i] = ld_frag(As + row * 40 + lh * 8, As + row * 40 + 16 + lh * 8);
    }
#pragma unroll
    for (int j = 0; j < 4; ++j) {
      int row = wn * 64 + j * 16 + l16;
      bf[j] = ld_frag(Bs + row * 40 + lh * 8, Bs + row * 40 + 16 + lh * 8);
    }
#pragma unroll
    for (int i = 0; i < 2; ++i)
#pragma unroll
      for (int j = 0; j < 4; ++j)
        acc[i][j] = wmma_f16(a[i], bf[j], acc[i][j]);
    __syncthreads();
  }

#pragma unroll
  for (int i = 0; i < 2; ++i)
#pragma unroll
    for (int j = 0; j < 4; ++j)
#pragma unroll
      for (int r = 0; r < 8; ++r) {
        int m = m0 + wm * 32 + i * 16 + r + 8 * lh;
        int n = n0 + wn * 64 + j * 16 + l16;
        out[(size_t)m * D_ + n] = acc[i][j][r];
      }
}

// ---------------------------------------------------------------------------
// Launch
// ---------------------------------------------------------------------------
extern "C" void kernel_launch(void* const* d_in, const int* in_sizes, int n_in,
                              void* d_out, int out_size, void* d_ws, size_t ws_size,
                              hipStream_t stream) {
  const float* x      = (const float*)d_in[0];
  const float* past_k = (const float*)d_in[1];
  const float* past_v = (const float*)d_in[2];
  const float* Wq     = (const float*)d_in[3];
  const float* Wk     = (const float*)d_in[4];
  const float* Wv     = (const float*)d_in[5];
  const float* Wo     = (const float*)d_in[6];
  float* out = (float*)d_out;

  char* ws = (char*)d_ws;
  float*     cosT = (float*)(ws);                          //  1 MB
  float*     sinT = (float*)(ws + (size_t)(1u << 20));     //  1 MB
  _Float16*  Qc   = (_Float16*)(ws + (size_t)(2u  << 20)); // 16 MB
  _Float16*  Kc   = (_Float16*)(ws + (size_t)(18u << 20)); // 32 MB
  _Float16*  Vt   = (_Float16*)(ws + (size_t)(50u << 20)); // 32 MB
  _Float16*  Yc   = (_Float16*)(ws + (size_t)(82u << 20)); // 16 MB

  k_rope_table<<<(S_ * HD_) / 256, 256, 0, stream>>>(cosT, sinT);
  k_past<<<dim3(P_ / 128, H_, B_), 256, 0, stream>>>(past_k, past_v, cosT, sinT, Kc, Vt);
  k_qkv<<<dim3(D_ / 128, (B_ * T_) / 128, 3), 256, 0, stream>>>(x, Wq, Wk, Wv,
                                                                cosT, sinT, Qc, Kc, Vt);
  k_attn<<<dim3(T_ / 128, H_, B_), 256, 0, stream>>>(Qc, Kc, Vt, Yc);
  k_oproj<<<dim3(D_ / 128, (B_ * T_) / 128), 256, 0, stream>>>(Yc, Wo, out);
}
